// TopKSparseAutoencoder_4071628997265
// MI455X (gfx1250) — matrix operand order repaired and checked by reading, not compiled, not run on hardware
//
#include <hip/hip_runtime.h>
#include <hip/hip_bf16.h>

// TopK Sparse Autoencoder for MI455X (gfx1250), wave32 + WMMA bf16 + TDM.
// B=8192, D=4096, H=16384, k=64.
// Pipeline:
//   1) x, W_enc  fp32 -> bf16 (once; bf16 W_enc fits the 192MB L2)
//   2) W_dec fp32 [D,H] -> bf16 transposed [H,D]
//   3) encoder GEMM: bf16 WMMA, fp32 acc, tiles DMA'd by tensor_load_to_lds
//   4) per-row exact top-64 (4-pass radix select on float bits)
//   5) sparse decoder: 64 coalesced bf16 row-gathers per output row

#define B_DIM 8192
#define D_DIM 4096
#define H_DIM 16384
#define K_TOP 64

typedef __attribute__((ext_vector_type(16))) __bf16 v16bf;
typedef __attribute__((ext_vector_type(8)))  __bf16 v8bf;
typedef __attribute__((ext_vector_type(8)))  float  v8f;
typedef __attribute__((ext_vector_type(4))) unsigned uint32x4;
typedef __attribute__((ext_vector_type(8))) int      int32x8;
typedef __attribute__((ext_vector_type(4))) int      int32x4;

#if __has_builtin(__builtin_amdgcn_tensor_load_to_lds) && \
    __has_builtin(__builtin_amdgcn_s_wait_tensorcnt)
#define USE_TDM 1
#endif
#if __has_include(<hip/amd_detail/amd_gfx1250_TDM.h>)
#define TDM_SIX_ARG 1   // amdgpu-toolchain (therock headers) => 6-arg builtin
#endif

// LDS tile: 128 rows x 32 bf16, row stride padded to 40 elems (80B):
// 16B-aligned fragment chunks, conflict-free ds_load_b128 across 16 lanes.
#define LSE 40

// ---------------------------------------------------------------------------
// TDM: DMA one 128x32 bf16 tile from a row-major [nrows, D_DIM] bf16 tensor
// into LDS, with pad_amount=4 dwords after every 16-dword (one-row) interval
// so the LDS image lands at row stride 40 elems. Issued by one wave only.
// ---------------------------------------------------------------------------
#ifdef USE_TDM
__device__ __forceinline__ void tdm_load_tile(
    const __bf16* gbase, unsigned row0, unsigned kk, unsigned nrows,
    __bf16* ldsdst)
{
    const unsigned long long ga = (unsigned long long)(size_t)gbase
        + 2ull * ((unsigned long long)row0 * D_DIM + kk);
    const unsigned lds = (unsigned)(size_t)ldsdst;   // low 32 bits = LDS offset

    uint32x4 g0;
    g0[0] = 1u;                                       // count=1, user mode
    g0[1] = lds;                                      // lds_addr
    g0[2] = (unsigned)(ga & 0xFFFFFFFFull);           // global_addr[31:0]
    g0[3] = (unsigned)((ga >> 32) & 0x1FFFFFFull)     // global_addr[56:32]
          | (2u << 30);                               // type=2 ("image")

    int32x8 g1;
    g1[0] = (int)((1u << 16)       // data_size = 2 bytes
                | (1u << 20)       // pad_enable
                | (3u << 22)       // pad_interval: 16 dwords (one tile row)
                | (3u << 25));     // pad_amount:   4 dwords
    g1[1] = (int)((unsigned)D_DIM << 16);             // tensor_dim0[15:0]
    g1[2] = (int)(((unsigned)D_DIM >> 16)             // tensor_dim0[31:16]
                | ((nrows & 0xFFFFu) << 16));         // tensor_dim1[15:0]
    g1[3] = (int)((nrows >> 16) | (32u << 16));       // dim1 hi | tile_dim0=32
    g1[4] = (int)(128u);                              // tile_dim1=128, tile_dim2=0
    g1[5] = (int)D_DIM;                               // tensor_dim0_stride
    g1[6] = 0;
    g1[7] = 0;

    const int32x4 z4 = {0, 0, 0, 0};
#ifdef TDM_SIX_ARG
    const int32x8 z8 = {0, 0, 0, 0, 0, 0, 0, 0};
    __builtin_amdgcn_tensor_load_to_lds(g0, g1, z4, z4, z8, 0);
#else
    __builtin_amdgcn_tensor_load_to_lds(g0, g1, z4, z4, 0);
#endif
}
#endif

// Manual fallback: cooperative bf16 tile copy (all 256 threads).
__device__ __forceinline__ void manual_load_tile(
    const __bf16* __restrict__ g, int row0, int kk, __bf16* lds, int tid)
{
#pragma unroll
    for (int it = 0; it < 2; ++it) {
        const int linear = it * 256 + tid;   // 512 chunks of 8 elems
        const int row = linear >> 2;         // 0..127
        const int c8  = (linear & 3) * 8;    // 0,8,16,24
        const v8bf v = *(const v8bf*)(g + (size_t)(row0 + row) * D_DIM + kk + c8);
        *(v8bf*)(lds + row * LSE + c8) = v;
    }
}

// One K=32 step: 4x2 WMMA per wave from padded LDS tiles.
__device__ __forceinline__ void compute_tile(
    const __bf16* Abuf, const __bf16* Bbuf,
    int waveM, int waveN, int half, int l15, v8f (&acc)[4][2])
{
    v16bf afrag[4];
#pragma unroll
    for (int mi = 0; mi < 4; ++mi) {
        const __bf16* p = Abuf + (waveM * 64 + mi * 16 + l15) * LSE;
        union { v16bf v; v8bf h[2]; } u;
        u.h[0] = *(const v8bf*)(p + half * 8);
        u.h[1] = *(const v8bf*)(p + 16 + half * 8);
        afrag[mi] = u.v;
    }
    v16bf bfrag[2];
#pragma unroll
    for (int ni = 0; ni < 2; ++ni) {
        const __bf16* p = Bbuf + (waveN * 32 + ni * 16 + l15) * LSE;
        union { v16bf v; v8bf h[2]; } u;
        u.h[0] = *(const v8bf*)(p + half * 8);
        u.h[1] = *(const v8bf*)(p + 16 + half * 8);
        bfrag[ni] = u.v;
    }
#pragma unroll
    for (int mi = 0; mi < 4; ++mi)
#pragma unroll
        for (int ni = 0; ni < 2; ++ni)
            acc[mi][ni] = __builtin_amdgcn_wmma_f32_16x16x32_bf16(
                false, afrag[mi], false, bfrag[ni],
                (short)0, acc[mi][ni], false, false);
}

// ---------------------------------------------------------------------------
// Encoder: features[b,h] = relu( sum_d x[b,d]*W_enc[h,d] + b_enc[h] )
// 128x128 tile/block, 8 waves (2M x 4N), double-buffered TDM tile loads.
// ---------------------------------------------------------------------------
__global__ __launch_bounds__(256) void encoder_kernel(
    const __bf16* __restrict__ xbf,   // [B, D] bf16
    const __bf16* __restrict__ wbf,   // [H, D] bf16
    const float*  __restrict__ benc,  // [H]
    float* __restrict__ feat)         // [B, H]
{
    __shared__ __bf16 Asm[2][128 * LSE];
    __shared__ __bf16 Bsm[2][128 * LSE];

    const int tid   = threadIdx.x;
    const int lane  = tid & 31;
    const int wave  = tid >> 5;
    const int waveM = wave & 1;
    const int waveN = wave >> 1;
    const int half  = lane >> 4;
    const int l15   = lane & 15;

    const int N0 = blockIdx.x * 128;  // over H
    const int M0 = blockIdx.y * 128;  // over B

    v8f acc[4][2];
#pragma unroll
    for (int mi = 0; mi < 4; ++mi)
#pragma unroll
        for (int ni = 0; ni < 2; ++ni)
#pragma unroll
            for (int e = 0; e < 8; ++e) acc[mi][ni][e] = 0.0f;

    const int T = D_DIM / 32;

#ifdef USE_TDM
    const bool issuer = (wave == 0);
    if (issuer) {
        tdm_load_tile(xbf, M0, 0, B_DIM, &Asm[0][0]);
        tdm_load_tile(wbf, N0, 0, H_DIM, &Bsm[0][0]);
    }
    for (int t = 0; t < T; ++t) {
        if (issuer) {
            if (t + 1 < T) {
                tdm_load_tile(xbf, M0, (t + 1) * 32, B_DIM, &Asm[(t + 1) & 1][0]);
                tdm_load_tile(wbf, N0, (t + 1) * 32, H_DIM, &Bsm[(t + 1) & 1][0]);
                __builtin_amdgcn_s_wait_tensorcnt(2);  // tile t complete
            } else {
                __builtin_amdgcn_s_wait_tensorcnt(0);
            }
        }
        __syncthreads();
        compute_tile(&Asm[t & 1][0], &Bsm[t & 1][0], waveM, waveN, half, l15, acc);
        __syncthreads();
    }
#else
    for (int t = 0; t < T; ++t) {
        manual_load_tile(xbf, M0, t * 32, &Asm[t & 1][0], tid);
        manual_load_tile(wbf, N0, t * 32, &Bsm[t & 1][0], tid);
        __syncthreads();
        compute_tile(&Asm[t & 1][0], &Bsm[t & 1][0], waveM, waveN, half, l15, acc);
        __syncthreads();
    }
#endif

    // C/D layout: VGPR r -> m = half*8 + r, n = lane&15. Bias + ReLU.
#pragma unroll
    for (int mi = 0; mi < 4; ++mi) {
#pragma unroll
        for (int ni = 0; ni < 2; ++ni) {
            const int gm = M0 + waveM * 64 + mi * 16 + half * 8;
            const int gn = N0 + waveN * 32 + ni * 16 + l15;
            const float bias = benc[gn];
#pragma unroll
            for (int r = 0; r < 8; ++r) {
                float v = acc[mi][ni][r] + bias;
                v = fmaxf(v, 0.0f);
                feat[(size_t)(gm + r) * H_DIM + gn] = v;
            }
        }
    }
}

// ---------------------------------------------------------------------------
// fp32 -> bf16 elementwise (8 elems/thread, exact-size grids).
// ---------------------------------------------------------------------------
__global__ __launch_bounds__(256) void f32_to_bf16_kernel(
    const float* __restrict__ src, __bf16* __restrict__ dst)
{
    const size_t i = ((size_t)blockIdx.x * 256 + threadIdx.x) * 8;
    const float4 a = *(const float4*)(src + i);
    const float4 b = *(const float4*)(src + i + 4);
    v8bf h;
    h[0] = (__bf16)a.x; h[1] = (__bf16)a.y; h[2] = (__bf16)a.z; h[3] = (__bf16)a.w;
    h[4] = (__bf16)b.x; h[5] = (__bf16)b.y; h[6] = (__bf16)b.z; h[7] = (__bf16)b.w;
    *(v8bf*)(dst + i) = h;
}

// ---------------------------------------------------------------------------
// Transpose W_dec [D,H] fp32 -> W_decT [H,D] bf16 (coalesced via LDS tile).
// ---------------------------------------------------------------------------
__global__ __launch_bounds__(256) void transpose_wdec_kernel(
    const float* __restrict__ Wdec, __bf16* __restrict__ WdecT)
{
    __shared__ float tile[32][33];
    const int tx = threadIdx.x & 31;
    const int ty = threadIdx.x >> 5;  // 0..7
    const int h0 = blockIdx.x * 32;
    const int d0 = blockIdx.y * 32;

#pragma unroll
    for (int i = 0; i < 4; ++i)
        tile[ty + i * 8][tx] = Wdec[(size_t)(d0 + ty + i * 8) * H_DIM + h0 + tx];
    __syncthreads();
#pragma unroll
    for (int i = 0; i < 4; ++i)
        WdecT[(size_t)(h0 + ty + i * 8) * D_DIM + d0 + tx] =
            (__bf16)tile[tx][ty + i * 8];
}

// ---------------------------------------------------------------------------
// Per-row exact top-64 via 4-pass 8-bit radix select on float bits
// (ReLU output >= 0 -> uint order == float order). Rewrites row in place,
// emits compacted (idx,val) pairs for the sparse decoder.
// ---------------------------------------------------------------------------
__global__ __launch_bounds__(256) void topk_kernel(
    float* __restrict__ feat, float* __restrict__ tvals, int* __restrict__ tidx)
{
    __shared__ int hist[256];
    __shared__ unsigned s_prefix;
    __shared__ int s_remaining;
    __shared__ int s_cnt, s_tie;

    const int row = blockIdx.x;
    const int tid = threadIdx.x;
    float* rp = feat + (size_t)row * H_DIM;

    if (tid == 0) { s_prefix = 0u; s_remaining = K_TOP; }

    for (int pass = 0; pass < 4; ++pass) {
        hist[tid] = 0;
        __syncthreads();
        const int shift = 24 - 8 * pass;
        const unsigned pref = s_prefix;
        for (int i = tid; i < H_DIM; i += 256) {
            const unsigned u = __float_as_uint(rp[i]);
            const bool ok = (pass == 0) || ((u >> (shift + 8)) == pref);
            if (ok) atomicAdd(&hist[(u >> shift) & 255], 1);
        }
        __syncthreads();
        if (tid == 0) {
            int c = 0, chosen = 0;
            for (int b = 255; b >= 0; --b) {
                const int nb = hist[b];
                if (c + nb >= s_remaining) { chosen = b; break; }
                c += nb;
            }
            s_remaining -= c;
            s_prefix = (s_prefix << 8) | (unsigned)chosen;
        }
        __syncthreads();
    }

    const unsigned T = s_prefix;     // bit pattern of 64th-largest value
    const int rkeep  = s_remaining;  // how many ==T survive
    if (tid == 0) { s_cnt = 0; s_tie = 0; }
    __syncthreads();

    for (int i = tid; i < H_DIM; i += 256) {
        const float v = rp[i];
        const unsigned u = __float_as_uint(v);
        if (u > T) {
            const int p = atomicAdd(&s_cnt, 1);
            tvals[row * K_TOP + p] = v;
            tidx[row * K_TOP + p] = i;
        } else if (u == T) {
            const int t = atomicAdd(&s_tie, 1);
            if (t < rkeep) {
                const int p = atomicAdd(&s_cnt, 1);
                tvals[row * K_TOP + p] = v;
                tidx[row * K_TOP + p] = i;
            } else {
                rp[i] = 0.0f;
            }
        } else {
            rp[i] = 0.0f;
        }
    }
}

// ---------------------------------------------------------------------------
// Sparse decoder: recon[b,d] = sum_{j<64} val_j * W_decT[idx_j, d]
// ---------------------------------------------------------------------------
__global__ __launch_bounds__(256) void decoder_kernel(
    const __bf16* __restrict__ WdecT,  // [H, D] bf16
    const float* __restrict__ tvals,
    const int* __restrict__ tidx,
    float* __restrict__ recon)          // [B, D]
{
    __shared__ float sv[K_TOP];
    __shared__ int   si[K_TOP];
    const int row = blockIdx.y;
    if (threadIdx.x < K_TOP) {
        sv[threadIdx.x] = tvals[row * K_TOP + threadIdx.x];
        si[threadIdx.x] = tidx[row * K_TOP + threadIdx.x];
    }
    __syncthreads();

    const int d = blockIdx.x * 256 + threadIdx.x;
    float acc = 0.0f;
#pragma unroll 8
    for (int j = 0; j < K_TOP; ++j) {
        const float w = (float)WdecT[(size_t)si[j] * D_DIM + d];
        acc = fmaf(sv[j], w, acc);
    }
    recon[(size_t)row * D_DIM + d] = acc;
}

// ---------------------------------------------------------------------------
extern "C" void kernel_launch(void* const* d_in, const int* in_sizes, int n_in,
                              void* d_out, int out_size, void* d_ws, size_t ws_size,
                              hipStream_t stream) {
    const float* x    = (const float*)d_in[0];  // [B, D]
    const float* Wenc = (const float*)d_in[1];  // [H, D]
    const float* benc = (const float*)d_in[2];  // [H]
    const float* Wdec = (const float*)d_in[3];  // [D, H]
    // d_in[4] = k (constant 64)

    float* feat  = (float*)d_out;                 // sparse_features [B, H]
    float* recon = feat + (size_t)B_DIM * H_DIM;  // reconstruction  [B, D]

    char* ws = (char*)d_ws;
    __bf16* xbf    = (__bf16*)ws;                                     //  64 MB
    __bf16* wencbf = (__bf16*)(ws + (size_t)B_DIM * D_DIM * 2);       // 128 MB
    __bf16* wdecT  = (__bf16*)(ws + (size_t)(B_DIM + H_DIM) * D_DIM * 2); // 128 MB
    float*  tvals  = (float*)(ws + (size_t)(B_DIM + 2 * H_DIM) * D_DIM * 2);
    int*    tidx   = (int*)(ws + (size_t)(B_DIM + 2 * H_DIM) * D_DIM * 2
                               + (size_t)B_DIM * K_TOP * 4);

    f32_to_bf16_kernel<<<(B_DIM * D_DIM) / 2048, 256, 0, stream>>>(x, xbf);
    f32_to_bf16_kernel<<<(H_DIM * D_DIM) / 2048, 256, 0, stream>>>(Wenc, wencbf);
    transpose_wdec_kernel<<<dim3(H_DIM / 32, D_DIM / 32), 256, 0, stream>>>(Wdec, wdecT);

    encoder_kernel<<<dim3(H_DIM / 128, B_DIM / 128), 256, 0, stream>>>(
        xbf, wencbf, benc, feat);
    topk_kernel<<<dim3(B_DIM), 256, 0, stream>>>(feat, tvals, tidx);
    decoder_kernel<<<dim3(D_DIM / 256, B_DIM), 256, 0, stream>>>(
        wdecT, tvals, tidx, recon);
}
